// DigitConvolutionalModel_3092376453552
// MI455X (gfx1250) — matrix operand order repaired and checked by reading, not compiled, not run on hardware
//
#include <hip/hip_runtime.h>

// ---------------- problem constants ----------------
#define BATCH   32768
#define IMG     28
#define OUTD    26          // 28 - 3 + 1
#define CONVN   (OUTD*OUTD) // 676
#define HID     200
#define NCLS    10

// ---------------- tiling ----------------
#define KPAD    704         // 676 padded to 22*32 (WMMA K=32 steps)
#define NPAD    224         // 200 padded to 14*16 (7 waves x 2 N-tiles)
#define KSTEPS  (KPAD/32)   // 22
#define MTILE   64          // batch rows per workgroup
#define NWAVES  7           // each wave owns two 16-wide HID tiles
#define NTHREADS (NWAVES*32)

// LDS layout: conv tile in bf16 (row stride padded for bank-conflict-free
// b128 A-fragment loads: 712*2B = 356 dwords, 356 % 64 = 36 = 4*9, 9 coprime 16),
// then h1 tile in f32.
#define CONV_STRIDE 712
#define H1_STRIDE   208
#define CONV_BYTES  (MTILE*CONV_STRIDE*2)
#define LDS_BYTES   (CONV_BYTES + MTILE*H1_STRIDE*4)

typedef __attribute__((ext_vector_type(16))) __bf16 v16bf;
typedef __attribute__((ext_vector_type(8)))  float  v8f;
typedef __attribute__((ext_vector_type(2)))  float  f32x2;

union BfFrag { uint4 q[2]; v16bf v; };

__device__ __forceinline__ unsigned short f2bf(float f) {
  unsigned int u = __float_as_uint(f);
  u += 0x7FFFu + ((u >> 16) & 1u);   // round-to-nearest-even
  return (unsigned short)(u >> 16);
}

// Pack W1 [200 x 676] f32 -> zero-padded bf16 [224 x 704] in workspace.
__global__ __launch_bounds__(256)
void pack_w1_bf16(const float* __restrict__ W1, unsigned short* __restrict__ W1b) {
  int idx = blockIdx.x * 256 + threadIdx.x;
  if (idx >= NPAD * KPAD) return;
  int n = idx / KPAD, k = idx % KPAD;
  float v = (n < HID && k < CONVN) ? W1[n * CONVN + k] : 0.0f;
  W1b[idx] = f2bf(v);
}

__global__ __launch_bounds__(NTHREADS)
void fused_conv_mlp(const float* __restrict__ x,
                    const float* __restrict__ convw,
                    const unsigned short* __restrict__ W1b,
                    const float* __restrict__ b1,
                    const float* __restrict__ W2,
                    const float* __restrict__ b2,
                    float* __restrict__ out) {
  extern __shared__ unsigned char smem[];
  unsigned short* convLds = (unsigned short*)smem;            // [MTILE][CONV_STRIDE] bf16
  float*          h1s     = (float*)(smem + CONV_BYTES);      // [MTILE][H1_STRIDE]  f32

  const int tid   = threadIdx.x;
  const int mBase = blockIdx.x * MTILE;

  // ---- zero the K padding region [676, 704) of every conv row (14 dwords/row)
  for (int idx = tid; idx < MTILE * 14; idx += NTHREADS) {
    int row = idx / 14, d = idx % 14;
    ((unsigned int*)(convLds + row * CONV_STRIDE + CONVN))[d] = 0u;
  }

  // ---- stage 1: conv (cross-correlation, valid) -> LDS bf16
  // packed-pair math so the backend can use v_pk_fma_f32 (2x fp32 FMA/instr)
  float w[9];
#pragma unroll
  for (int t = 0; t < 9; ++t) w[t] = convw[t];

  for (int task = tid; task < MTILE * OUTD; task += NTHREADS) {
    int m = task / OUTD;
    int i = task % OUTD;
    const float* xrow = x + (size_t)(mBase + m) * (IMG * IMG) + i * IMG;
    float xr[3][IMG];
#pragma unroll
    for (int a = 0; a < 3; ++a) {
      const float4* rp = (const float4*)(xrow + a * IMG);   // 16B aligned (28 floats/row)
#pragma unroll
      for (int q = 0; q < 7; ++q) {
        float4 v = rp[q];
        xr[a][4 * q + 0] = v.x; xr[a][4 * q + 1] = v.y;
        xr[a][4 * q + 2] = v.z; xr[a][4 * q + 3] = v.w;
      }
    }
    unsigned int* dst = (unsigned int*)(convLds + m * CONV_STRIDE + i * OUTD);
#pragma unroll
    for (int t = 0; t < 13; ++t) {
      const int j = 2 * t;
      f32x2 o = {0.0f, 0.0f};
#pragma unroll
      for (int a = 0; a < 3; ++a)
#pragma unroll
        for (int bcol = 0; bcol < 3; ++bcol) {
          f32x2 xp = { xr[a][j + bcol], xr[a][j + bcol + 1] };
          f32x2 wv = { w[a * 3 + bcol], w[a * 3 + bcol] };
          o = __builtin_elementwise_fma(xp, wv, o);         // v_pk_fma_f32
        }
      dst[t] = (unsigned int)f2bf(o.x) | ((unsigned int)f2bf(o.y) << 16);
    }
  }

  __syncthreads();

  // ---- stage 2: FC1 via v_wmma_f32_16x16x32_bf16
  // wave w computes h1[:, 32w .. 32w+31] (two 16-wide N tiles) for all 4
  // 16-row M subtiles: each A fragment feeds 2 wmmas -> half the LDS traffic.
  const int wave  = tid >> 5;
  const int lane  = tid & 31;
  const int nloc  = lane & 15;   // N column within tile; also A-fragment M row
  const int halfB = lane >> 4;   // lane half selects K sub-range per ISA layout
  const int n0    = wave * 32 + nloc;
  const int n1    = n0 + 16;

  v8f acc0[4] = {};
  v8f acc1[4] = {};

  const unsigned short* wrow0 = W1b + (size_t)n0 * KPAD;
  const unsigned short* wrow1 = W1b + (size_t)n1 * KPAD;

  for (int kt = 0; kt < KSTEPS; ++kt) {
    const int k0 = kt * 32;

    // B fragments: lane holds column n, 16 consecutive K at k0 + 16*halfB
    BfFrag bf0, bf1;
    {
      const uint4* bp0 = (const uint4*)(wrow0 + k0 + 16 * halfB);
      const uint4* bp1 = (const uint4*)(wrow1 + k0 + 16 * halfB);
      bf0.q[0] = bp0[0]; bf0.q[1] = bp0[1];
      bf1.q[0] = bp1[0]; bf1.q[1] = bp1[1];
    }
    if (kt + 1 < KSTEPS) {
      __builtin_prefetch(wrow0 + k0 + 32 + 16 * halfB, 0, 3);  // global_prefetch_b8
      __builtin_prefetch(wrow1 + k0 + 32 + 16 * halfB, 0, 3);
    }

#pragma unroll
    for (int s = 0; s < 4; ++s) {
      // A fragment (contiguous K to match B): lanes 0-15 K[0..15],
      // lanes 16-31 K[16..31]
      const unsigned short* arow = convLds + (s * 16 + nloc) * CONV_STRIDE;
      BfFrag af;
      af.q[0] = *(const uint4*)(arow + k0 + 16 * halfB);
      af.q[1] = *(const uint4*)(arow + k0 + 16 * halfB + 8);
      acc0[s] = __builtin_amdgcn_wmma_f32_16x16x32_bf16(
          false, af.v, false, bf0.v, (short)0, acc0[s], false, false);
      acc1[s] = __builtin_amdgcn_wmma_f32_16x16x32_bf16(
          false, af.v, false, bf1.v, (short)0, acc1[s], false, false);
    }
  }

  // bias + ReLU, scatter to LDS h1 (C/D layout: VGPR r, lane -> M = r + 8*halfB, N = nloc)
  if (n0 < HID) {
    const float bias = b1[n0];
#pragma unroll
    for (int s = 0; s < 4; ++s)
#pragma unroll
      for (int r = 0; r < 8; ++r) {
        float v = acc0[s][r] + bias;
        h1s[(s * 16 + r + 8 * halfB) * H1_STRIDE + n0] = fmaxf(v, 0.0f);
      }
  }
  if (n1 < HID) {
    const float bias = b1[n1];
#pragma unroll
    for (int s = 0; s < 4; ++s)
#pragma unroll
      for (int r = 0; r < 8; ++r) {
        float v = acc1[s][r] + bias;
        h1s[(s * 16 + r + 8 * halfB) * H1_STRIDE + n1] = fmaxf(v, 0.0f);
      }
  }

  __syncthreads();

  // ---- stage 3: FC2 (64x10, K=200) on VALU; W2/b2 are L2-resident
  for (int t = tid; t < MTILE * NCLS; t += NTHREADS) {
    int m = t / NCLS, c = t % NCLS;
    const float* h  = h1s + m * H1_STRIDE;
    const float* wr = W2 + c * HID;
    float s0 = 0.f, s1 = 0.f, s2 = 0.f, s3 = 0.f;
    for (int k = 0; k < HID; k += 4) {
      s0 = fmaf(h[k + 0], wr[k + 0], s0);
      s1 = fmaf(h[k + 1], wr[k + 1], s1);
      s2 = fmaf(h[k + 2], wr[k + 2], s2);
      s3 = fmaf(h[k + 3], wr[k + 3], s3);
    }
    out[(size_t)(mBase + m) * NCLS + c] = b2[c] + ((s0 + s1) + (s2 + s3));
  }
}

extern "C" void kernel_launch(void* const* d_in, const int* in_sizes, int n_in,
                              void* d_out, int out_size, void* d_ws, size_t ws_size,
                              hipStream_t stream) {
  (void)in_sizes; (void)n_in; (void)out_size; (void)ws_size;
  const float* x     = (const float*)d_in[0];
  const float* convw = (const float*)d_in[1];
  const float* W1    = (const float*)d_in[2];
  const float* b1    = (const float*)d_in[3];
  const float* W2    = (const float*)d_in[4];
  const float* b2    = (const float*)d_in[5];
  float* out = (float*)d_out;

  // d_ws: bf16 packed, zero-padded W1 [224 x 704] = 315,392 bytes
  unsigned short* W1b = (unsigned short*)d_ws;

  const int packN = NPAD * KPAD;
  pack_w1_bf16<<<(packN + 255) / 256, 256, 0, stream>>>(W1, W1b);

  fused_conv_mlp<<<BATCH / MTILE, NTHREADS, LDS_BYTES, stream>>>(
      x, convw, W1b, b1, W2, b2, out);
}
